// MoE_88493506167191
// MI455X (gfx1250) — compile-verified, matrix operand order checked by reading, not run on hardware
//
#include <hip/hip_runtime.h>
#include <hip/hip_bf16.h>

// ---------------- problem constants (from reference) ----------------
#define T_TOKENS 4096   // B*S = 2*2048
#define DIM      1024   // D
#define FDIM     4096   // F
#define NEXP     8      // experts
#define FC       128    // F-chunk held in LDS between the two GEMMs

// W1 tiling: K=DIM (32 k-tiles), N=FDIM (256 n-tiles)
#define W1_KT 32
#define W1_NT 256
// W2 tiling: K=FDIM (128 k-tiles), N=DIM (64 n-tiles)
#define W2_KT 128
#define W2_NT 64

typedef __attribute__((ext_vector_type(16))) __bf16 v16bf;
typedef __attribute__((ext_vector_type(8)))  float  v8f;

union Frag32 {                 // one lane's 32-byte WMMA operand fragment
    v16bf          v;
    uint4          q[2];
    unsigned int   d[8];
    unsigned short u[16];
};

// 16-bit A/B fragment element -> K index (ISA 7.12.2, 16-bit matrix 16x32):
// element pair (2j,2j+1): j<4 -> K = 2j+p (+8 for lanes 16..31),
//                         j>=4 -> K = 16+2(j-4)+p (+8 for lanes 16..31)
__device__ __forceinline__ int frag_k(int i, int half) {
    int j = i >> 1, p = i & 1;
    int k = (j < 4) ? (2 * j + p) : (8 + 2 * j + p);
    return k + 8 * half;
}

__device__ __forceinline__ unsigned short f2bf(float f) {
    unsigned int u = __builtin_bit_cast(unsigned int, f);
    unsigned int r = u + 0x7FFFu + ((u >> 16) & 1u);   // round-to-nearest-even
    return (unsigned short)(r >> 16);
}

__device__ __forceinline__ v8f zero8() {
    v8f z;
#pragma unroll
    for (int i = 0; i < 8; ++i) z[i] = 0.0f;
    return z;
}

// ---------------- kernel 0: zero d_out and routing counts ----------------
__global__ void moe_init_kernel(float* __restrict__ out, int n, int* __restrict__ counts) {
    int i = blockIdx.x * blockDim.x + threadIdx.x;
    if (i < n) out[i] = 0.0f;
    if (i < NEXP) counts[i] = 0;
}

// ---------------- kernel 1: fp32 -> bf16 row-major conversion (for x) ----------------
__global__ void convert_bf16_kernel(const float* __restrict__ src,
                                    unsigned short* __restrict__ dst, int n4) {
    int i = blockIdx.x * blockDim.x + threadIdx.x;
    if (i >= n4) return;
    float4 f = ((const float4*)src)[i];
    ushort4 o;
    o.x = f2bf(f.x); o.y = f2bf(f.y); o.z = f2bf(f.z); o.w = f2bf(f.w);
    ((ushort4*)dst)[i] = o;
}

// ---------------- kernel 2: fp32 weights -> bf16 WMMA-B-fragment order ----------------
// One wave32 per 32x16 (KxN) tile. dst layout: [e][kt][nt][lane][16] bf16, so each
// lane's GEMM operand is a contiguous 32-byte chunk (two global_load_b128 in the FFN).
__global__ void swizzle_w_kernel(const float* __restrict__ src,
                                 unsigned short* __restrict__ dst,
                                 int K, int N, int ntilesN, int tiles_per_mat) {
    int gw   = (blockIdx.x * blockDim.x + threadIdx.x) >> 5;   // global wave = tile id
    int lane = threadIdx.x & 31;
    if (gw >= NEXP * tiles_per_mat) return;
    int e = gw / tiles_per_mat;
    int t = gw % tiles_per_mat;
    int kt = t / ntilesN;
    int nt = t % ntilesN;
    const float* s = src + (size_t)e * K * N;
    int half = lane >> 4, nl = lane & 15;
    Frag32 f;
#pragma unroll
    for (int i = 0; i < 16; ++i) {
        int k = kt * 32 + frag_k(i, half);
        f.u[i] = f2bf(s[(size_t)k * N + nt * 16 + nl]);
    }
    unsigned short* d = dst + ((size_t)gw * 32 + lane) * 16;
    ((uint4*)d)[0] = f.q[0];
    ((uint4*)d)[1] = f.q[1];
}

// ---------------- kernel 3: noisy top-2 router (one wave32 per token) ----------------
__global__ void router_kernel(const float* __restrict__ x,
                              const float* __restrict__ noise,
                              const float* __restrict__ Wr, const float* __restrict__ br,
                              const float* __restrict__ Wn, const float* __restrict__ bn,
                              int* __restrict__ counts,
                              int* __restrict__ toks, float* __restrict__ gates) {
    int wid  = (blockIdx.x * blockDim.x + threadIdx.x) >> 5;   // token id
    int lane = threadIdx.x & 31;
    if (wid >= T_TOKENS) return;

    const float* xr = x + (size_t)wid * DIM;
    float acc[NEXP], nacc[NEXP];
#pragma unroll
    for (int e = 0; e < NEXP; ++e) { acc[e] = 0.0f; nacc[e] = 0.0f; }

    for (int d = lane; d < DIM; d += 32) {
        float xv = xr[d];
        const float* wr = Wr + (size_t)d * NEXP;
        const float* wn = Wn + (size_t)d * NEXP;
#pragma unroll
        for (int e = 0; e < NEXP; ++e) {
            acc[e]  = fmaf(xv, wr[e], acc[e]);
            nacc[e] = fmaf(xv, wn[e], nacc[e]);
        }
    }
#pragma unroll
    for (int e = 0; e < NEXP; ++e) {
        for (int off = 16; off > 0; off >>= 1) {
            acc[e]  += __shfl_xor(acc[e],  off, 32);
            nacc[e] += __shfl_xor(nacc[e], off, 32);
        }
    }

    if (lane == 0) {
        float noisy[NEXP];
#pragma unroll
        for (int e = 0; e < NEXP; ++e) {
            float lg = acc[e] + br[e];
            float nl = nacc[e] + bn[e];
            float sp = fmaxf(nl, 0.0f) + log1pf(expf(-fabsf(nl)));  // softplus
            noisy[e] = lg + noise[(size_t)wid * NEXP + e] * sp;
        }
        int i0 = 0;
#pragma unroll
        for (int e = 1; e < NEXP; ++e) if (noisy[e] > noisy[i0]) i0 = e;
        int i1 = (i0 == 0) ? 1 : 0;
#pragma unroll
        for (int e = 0; e < NEXP; ++e)
            if (e != i0 && noisy[e] > noisy[i1]) i1 = e;

        float v0 = noisy[i0], v1 = noisy[i1];
        float p1 = 1.0f / (1.0f + expf(v0 - v1));   // 2-way softmax
        float p0 = 1.0f - p1;

        int pos0 = atomicAdd(&counts[i0], 1);
        toks [i0 * T_TOKENS + pos0] = wid;
        gates[i0 * T_TOKENS + pos0] = p0;
        int pos1 = atomicAdd(&counts[i1], 1);
        toks [i1 * T_TOKENS + pos1] = wid;
        gates[i1 * T_TOKENS + pos1] = p1;
    }
}

// ---------------- kernel 4: routed expert FFN with bf16 WMMA ----------------
// grid = (256 token-blocks, 8 experts), block = 256 threads (8 wave32s).
__global__ __launch_bounds__(256)
void ffn_kernel(const unsigned short* __restrict__ xb,
                const unsigned short* __restrict__ w1s,   // fragment-order W1
                const unsigned short* __restrict__ w2s,   // fragment-order W2
                const float* __restrict__ b1, const float* __restrict__ b2,
                const int* __restrict__ counts,
                const int* __restrict__ toks, const float* __restrict__ gatesL,
                float* __restrict__ out) {
    int e  = blockIdx.y;
    int tb = blockIdx.x;
    int cnt  = counts[e];
    int base = tb * 16;
    if (base >= cnt) return;                        // empty block: early exit

    __shared__ int   s_tok[16];
    __shared__ float s_gate[16];
    // A-fragment-order LDS: [kt][lane][16] bf16
    __shared__ __align__(16) unsigned short xsf[W1_KT * 32 * 16];  // 32 KB
    __shared__ __align__(16) unsigned short hsf[4 * 32 * 16];      // 4 KB (FC/32 k-tiles)

    int tid = threadIdx.x;
    if (tid < 16) {
        int r = base + tid;
        if (r < cnt) {
            s_tok[tid]  = toks  [e * T_TOKENS + r];
            s_gate[tid] = gatesL[e * T_TOKENS + r];
        } else {                                    // pad rows: valid token, zero gate
            s_tok[tid]  = toks[e * T_TOKENS + base];
            s_gate[tid] = 0.0f;
        }
    }
    __syncthreads();

    // gather x rows straight into A-fragment order (pairs of K are contiguous -> u32 loads)
    for (int pair = tid; pair < W1_KT * 32; pair += 256) {   // (kt, lane') pairs
        int kt = pair >> 5;
        int l  = pair & 31;
        int hf = l >> 4, m = l & 15;
        const unsigned short* src = xb + (size_t)s_tok[m] * DIM + kt * 32;
        Frag32 f;
#pragma unroll
        for (int j = 0; j < 8; ++j)
            f.d[j] = *(const unsigned int*)(src + frag_k(2 * j, hf));
        uint4* d = (uint4*)&xsf[(size_t)pair * 16];
        d[0] = f.q[0];
        d[1] = f.q[1];
    }
    __syncthreads();

    int wave = tid >> 5;
    int lane = tid & 31;
    int half = lane >> 4;
    int nl   = lane & 15;
    int mbase = half * 8;                // C/D layout: VGPR v -> M = v + 8*half

    // h-writer inverse fragment mapping (per-lane constants):
    // this wave's h tile covers GEMM2-K columns kloc = wave*16 + nl (mod 32 within k-tile)
    int kt_h = wave >> 1;
    int kloc = ((wave & 1) << 4) | nl;
    int hfp  = (kloc >> 3) & 1;
    int kk0  = kloc - 8 * hfp;
    int ii   = (kk0 < 8) ? kk0 : (kk0 - 8);

    const unsigned short* w1e = w1s + (size_t)e * W1_KT * W1_NT * 512;
    const unsigned short* w2e = w2s + (size_t)e * W2_KT * W2_NT * 512;
    const float* b1e = b1 + (size_t)e * FDIM;
    const float* b2e = b2 + (size_t)e * DIM;

    v8f oacc[8];                        // out[16,1024]: 8 n-tiles per wave
#pragma unroll
    for (int j = 0; j < 8; ++j) oacc[j] = zero8();

    for (int fo = 0; fo < FDIM; fo += FC) {
        // -------- GEMM1: h tile [16 x 16] at cols fc0, K-loop over D --------
        int fc0 = fo + wave * 16;
        int nt1 = fc0 >> 4;
        const unsigned short* w1p = w1e + ((size_t)nt1 * 32 + lane) * 16;
        const size_t w1_kstride = (size_t)W1_NT * 512;       // u16 per k-tile
        v8f hacc = zero8();
#pragma unroll 4
        for (int kt = 0; kt < W1_KT; ++kt) {
            Frag32 a, b;
            const uint4* ap = (const uint4*)&xsf[(size_t)(kt * 32 + lane) * 16];
            a.q[0] = ap[0]; a.q[1] = ap[1];
            const uint4* bp = (const uint4*)(w1p + (size_t)kt * w1_kstride);
            b.q[0] = bp[0]; b.q[1] = bp[1];
            __builtin_prefetch(w1p + (size_t)(kt + 4) * w1_kstride, 0, 1);
            hacc = __builtin_amdgcn_wmma_f32_16x16x32_bf16(
                false, a.v, false, b.v, (short)0, hacc, false, false);
        }
        // bias + relu, scatter into A-fragment-order hsf
        {
            float bias = b1e[fc0 + nl];
#pragma unroll
            for (int v = 0; v < 8; ++v) {
                float hv = fmaxf(hacc[v] + bias, 0.0f);
                int lp = (v + mbase) + 16 * hfp;             // lane' holding row m
                hsf[(size_t)(kt_h * 32 + lp) * 16 + ii] = f2bf(hv);
            }
        }
        __syncthreads();

        // -------- GEMM2: out n-tiles += h[16,FC] @ W2[fo:fo+FC, :] --------
        int ktF0 = fo >> 5;
        const size_t w2_kstride = (size_t)W2_NT * 512;       // u16 per k-tile
#pragma unroll
        for (int j = 0; j < 8; ++j) {
            int ntD = wave * 8 + j;
            const unsigned short* w2p =
                w2e + ((size_t)ktF0 * W2_NT + ntD) * 512 + (size_t)lane * 16;
            __builtin_prefetch(w2p + 4 * w2_kstride, 0, 1);
#pragma unroll
            for (int kk = 0; kk < 4; ++kk) {
                Frag32 a, b;
                const uint4* ap = (const uint4*)&hsf[(size_t)(kk * 32 + lane) * 16];
                a.q[0] = ap[0]; a.q[1] = ap[1];
                const uint4* bp = (const uint4*)(w2p + (size_t)kk * w2_kstride);
                b.q[0] = bp[0]; b.q[1] = bp[1];
                oacc[j] = __builtin_amdgcn_wmma_f32_16x16x32_bf16(
                    false, a.v, false, b.v, (short)0, oacc[j], false, false);
            }
        }
        __syncthreads();   // hsf reused next chunk
    }

    // -------- gate-weighted scatter: out[tok] += gate * (acc + b2) --------
#pragma unroll
    for (int j = 0; j < 8; ++j) {
        int n = (wave * 8 + j) * 16 + nl;
        float bias = b2e[n];
#pragma unroll
        for (int v = 0; v < 8; ++v) {
            int m = mbase + v;
            float val = s_gate[m] * (oacc[j][v] + bias);
            atomicAdd(&out[(size_t)s_tok[m] * DIM + n], val);
        }
    }
}

// ---------------- host-side launcher ----------------
extern "C" void kernel_launch(void* const* d_in, const int* in_sizes, int n_in,
                              void* d_out, int out_size, void* d_ws, size_t ws_size,
                              hipStream_t stream) {
    const float* x     = (const float*)d_in[0];
    const float* noise = (const float*)d_in[1];
    const float* Wr    = (const float*)d_in[2];
    const float* br    = (const float*)d_in[3];
    const float* Wn    = (const float*)d_in[4];
    const float* bn    = (const float*)d_in[5];
    const float* W1    = (const float*)d_in[6];
    const float* b1    = (const float*)d_in[7];
    const float* W2    = (const float*)d_in[8];
    const float* b2    = (const float*)d_in[9];
    float* out = (float*)d_out;

    // workspace layout
    char*  ws     = (char*)d_ws;
    size_t off    = 0;
    int*   counts = (int*)(ws + off);               off += 256;
    int*   toks   = (int*)(ws + off);               off += (size_t)NEXP * T_TOKENS * sizeof(int);
    float* gates  = (float*)(ws + off);             off += (size_t)NEXP * T_TOKENS * sizeof(float);
    off = (off + 255) & ~(size_t)255;
    unsigned short* xb  = (unsigned short*)(ws + off); off += (size_t)T_TOKENS * DIM * 2;
    unsigned short* w1s = (unsigned short*)(ws + off); off += (size_t)NEXP * DIM * FDIM * 2;
    unsigned short* w2s = (unsigned short*)(ws + off); off += (size_t)NEXP * FDIM * DIM * 2;
    (void)ws_size; (void)in_sizes; (void)n_in;

    // 0) zero output + routing counts
    {
        int n = out_size;   // 4096*1024
        moe_init_kernel<<<(n + 255) / 256, 256, 0, stream>>>(out, n, counts);
    }
    // 1) x: fp32 -> bf16 row-major
    {
        int n4 = (T_TOKENS * DIM) / 4;
        convert_bf16_kernel<<<(n4 + 255) / 256, 256, 0, stream>>>(x, xb, n4);
    }
    // 2) W1/W2: fp32 -> bf16 pre-swizzled into WMMA B-fragment order (L2-resident)
    {
        int tiles1 = W1_KT * W1_NT;      // 8192 tiles per expert
        int waves1 = NEXP * tiles1;
        swizzle_w_kernel<<<(waves1 * 32 + 255) / 256, 256, 0, stream>>>(
            W1, w1s, DIM, FDIM, W1_NT, tiles1);
        int tiles2 = W2_KT * W2_NT;      // 8192 tiles per expert
        int waves2 = NEXP * tiles2;
        swizzle_w_kernel<<<(waves2 * 32 + 255) / 256, 256, 0, stream>>>(
            W2, w2s, FDIM, DIM, W2_NT, tiles2);
    }
    // 3) router: one wave32 per token, builds per-expert token/gate lists
    {
        int threads = T_TOKENS * 32;
        router_kernel<<<threads / 256, 256, 0, stream>>>(x, noise, Wr, br, Wn, bn,
                                                         counts, toks, gates);
    }
    // 4) routed expert FFN (bf16 WMMA) + gated atomic combine
    {
        dim3 grid(T_TOKENS / 16, NEXP);  // worst case: every token on one expert
        ffn_kernel<<<grid, 256, 0, stream>>>(xb, w1s, w2s, b1, b2,
                                             counts, toks, gates, out);
    }
}
